// MultilayerGRU_61607010893843
// MI455X (gfx1250) — compile-verified, matrix operand order checked by reading, not run on hardware
//
#include <hip/hip_runtime.h>
#include <hip/hip_bf16.h>

typedef __attribute__((ext_vector_type(16))) _Float16 v16h;
typedef __attribute__((ext_vector_type(8)))  _Float16 v8h;
typedef __attribute__((ext_vector_type(8)))  float    v8f;

#define GB   32
#define GS   512
#define GI   1024
#define GH   1024
#define GO   1024
#define GL   2

// ---------------------------------------------------------------------------
// Fragment loader: 16 f16 values for one lane, per the CDNA5 16-bit A/B VGPR
// layout (ISA 7.12.2):
//   lanes 0-15 : halves 0..7 -> K = k+0..7,   halves 8..15 -> K = k+16..23
//   lanes 16-31: halves 0..7 -> K = k+8..15,  halves 8..15 -> K = k+24..31
// ---------------------------------------------------------------------------
__device__ __forceinline__ v16h load_frag16(const _Float16* __restrict__ base,
                                            long row_off, long k, int ksel) {
  const _Float16* p = base + row_off + k + ksel;
  v8h lo = *(const v8h*)(p);
  v8h hi = *(const v8h*)(p + 16);
  return __builtin_shufflevector(lo, hi, 0,1,2,3,4,5,6,7,8,9,10,11,12,13,14,15);
}

__device__ __forceinline__ v8f wmma_f16(v16h a, v16h b, v8f c) {
  return __builtin_amdgcn_wmma_f32_16x16x32_f16(false, a, false, b, (short)0, c, false, false);
}

// cMN: M-tile (rows 0..15 / 16..31), N-tile (cols n0.. / n0+16..)
struct Acc4 { v8f c00, c10, c01, c11; };

// ---------------------------------------------------------------------------
// One 32x32 output slab of  C = [Alo | Ahi] @ W^T  (W row-major [N, K1+K2]).
// A fragments are shared across the two N-tiles and B fragments across the
// two M-tiles: 8 b128 loads feed 4 WMMAs per K-step (2 loads per WMMA).
// ---------------------------------------------------------------------------
__device__ __forceinline__ Acc4 gemm_slab(const _Float16* __restrict__ Alo, long ldalo, int K1,
                                          const _Float16* __restrict__ Ahi, long ldahi, int K2,
                                          const _Float16* __restrict__ W, long ldw,
                                          int n0, int lane) {
  const int row  = lane & 15;
  const int ksel = (lane >> 4) << 3;   // 0 or 8
  v8f c00 = {}, c10 = {}, c01 = {}, c11 = {};
  const long a0lo = (long)row        * ldalo;
  const long a1lo = (long)(row + 16) * ldalo;
  const long a0hi = (long)row        * ldahi;
  const long a1hi = (long)(row + 16) * ldahi;
  const _Float16* Wrow0 = W + (long)(n0 + row)      * ldw;
  const _Float16* Wrow1 = W + (long)(n0 + 16 + row) * ldw;

#pragma unroll 4
  for (int k = 0; k < K1; k += 32) {
    v16h b0 = load_frag16(Wrow0, 0,    k, ksel);
    v16h b1 = load_frag16(Wrow1, 0,    k, ksel);
    v16h a0 = load_frag16(Alo,   a0lo, k, ksel);
    v16h a1 = load_frag16(Alo,   a1lo, k, ksel);
    c00 = wmma_f16(a0, b0, c00);
    c10 = wmma_f16(a1, b0, c10);
    c01 = wmma_f16(a0, b1, c01);
    c11 = wmma_f16(a1, b1, c11);
  }
#pragma unroll 4
  for (int k = 0; k < K2; k += 32) {
    v16h b0 = load_frag16(Wrow0, 0,    K1 + k, ksel);
    v16h b1 = load_frag16(Wrow1, 0,    K1 + k, ksel);
    v16h a0 = load_frag16(Ahi,   a0hi, k, ksel);
    v16h a1 = load_frag16(Ahi,   a1hi, k, ksel);
    c00 = wmma_f16(a0, b0, c00);
    c10 = wmma_f16(a1, b0, c10);
    c01 = wmma_f16(a0, b1, c01);
    c11 = wmma_f16(a1, b1, c11);
  }
  return {c00, c10, c01, c11};
}

__device__ __forceinline__ float sigmoid_f(float x) {
  return 1.0f / (1.0f + __expf(-x));
}

__device__ __forceinline__ v8f acc_sel(const Acc4& a, int mt, int nt) {
  return mt ? (nt ? a.c11 : a.c10) : (nt ? a.c01 : a.c00);
}

// C/D element i maps to m = 16*mt + i + 8*(lane>=16), n = n0 + 16*nt + (lane&15)

// ---------------------------------------------------------------------------
// z/r gate GEMM:  pre = [inp | h] @ [Wz|Wr packed]^T ; N = 2H
//   n <  H : Z[m,n]      = sigmoid(pre + bz[n])
//   n >= H : rh16[m,n-H] = f16( sigmoid(pre + br[n-H]) * h[m,n-H] )
// grid: 16 blocks x 128 thr (4 waves) = 64 slabs of 32 cols
// ---------------------------------------------------------------------------
__global__ void gru_zr_kernel(const _Float16* __restrict__ Alo, long ldalo,
                              const _Float16* __restrict__ Ahi,
                              const _Float16* __restrict__ Wzr,
                              const float* __restrict__ bz, const float* __restrict__ br,
                              const float* __restrict__ hf,
                              float* __restrict__ Z, _Float16* __restrict__ rh16) {
  const int lane = threadIdx.x & 31;
  const int n0   = (blockIdx.x * 4 + (threadIdx.x >> 5)) * 32;
  Acc4 acc = gemm_slab(Alo, ldalo, GI, Ahi, GH, GH, Wzr, 2 * GH, n0, lane);

  const int mb = (lane >> 4) << 3;
#pragma unroll
  for (int nt = 0; nt < 2; ++nt) {
    const int n = n0 + 16 * nt + (lane & 15);
#pragma unroll
    for (int mt = 0; mt < 2; ++mt) {
      const v8f c = acc_sel(acc, mt, nt);
#pragma unroll
      for (int i = 0; i < 8; ++i) {
        const int m = 16 * mt + mb + i;
        if (n < GH) {
          Z[m * GH + n] = sigmoid_f(c[i] + bz[n]);
        } else {
          const int nn = n - GH;
          const float r = sigmoid_f(c[i] + br[nn]);
          rh16[m * GH + nn] = (_Float16)(r * hf[m * GH + nn]);
        }
      }
    }
  }
}

// ---------------------------------------------------------------------------
// g gate GEMM + state update:  pre = [inp | r*h] @ [Wgx|Wgh]^T ; N = H
//   g = tanh(pre + bg);  h_new = (1-z)h + z*g ; write h f32 + f16 copy
// grid: 8 blocks x 128 thr = 32 slabs
// ---------------------------------------------------------------------------
__global__ void gru_g_kernel(const _Float16* __restrict__ Alo, long ldalo,
                             const _Float16* __restrict__ rh16,
                             const _Float16* __restrict__ Wg,
                             const float* __restrict__ bg,
                             const float* __restrict__ Z,
                             float* __restrict__ hf, _Float16* __restrict__ h16out) {
  const int lane = threadIdx.x & 31;
  const int n0   = (blockIdx.x * 4 + (threadIdx.x >> 5)) * 32;
  Acc4 acc = gemm_slab(Alo, ldalo, GI, rh16, GH, GH, Wg, 2 * GH, n0, lane);

  const int mb = (lane >> 4) << 3;
#pragma unroll
  for (int nt = 0; nt < 2; ++nt) {
    const int n = n0 + 16 * nt + (lane & 15);
#pragma unroll
    for (int mt = 0; mt < 2; ++mt) {
      const v8f c = acc_sel(acc, mt, nt);
#pragma unroll
      for (int i = 0; i < 8; ++i) {
        const int m   = 16 * mt + mb + i;
        const int idx = m * GH + n;
        const float g  = tanhf(c[i] + bg[n]);
        const float zv = Z[idx];
        const float hn = (1.0f - zv) * hf[idx] + zv * g;
        hf[idx]     = hn;
        h16out[idx] = (_Float16)hn;
      }
    }
  }
}

// ---------------------------------------------------------------------------
// Output projection: y[:,t,:] = h16 @ Wo^T + bo ; grid 8 x 128 = 32 slabs
// ---------------------------------------------------------------------------
__global__ void gru_out_kernel(const _Float16* __restrict__ A,
                               const _Float16* __restrict__ Wo16,
                               const float* __restrict__ bo,
                               float* __restrict__ y, int t) {
  const int lane = threadIdx.x & 31;
  const int n0   = (blockIdx.x * 4 + (threadIdx.x >> 5)) * 32;
  Acc4 acc = gemm_slab(A, GH, GH, A, GH, 0, Wo16, GH, n0, lane);

  const int mb = (lane >> 4) << 3;
#pragma unroll
  for (int nt = 0; nt < 2; ++nt) {
    const int n = n0 + 16 * nt + (lane & 15);
#pragma unroll
    for (int mt = 0; mt < 2; ++mt) {
      const v8f c = acc_sel(acc, mt, nt);
#pragma unroll
      for (int i = 0; i < 8; ++i) {
        const int m = 16 * mt + mb + i;
        y[((long)m * GS + t) * GO + n] = c[i] + bo[n];
      }
    }
  }
}

// ------------------------- prep / epilogue kernels -------------------------
__global__ void conv_f16_kernel(_Float16* __restrict__ dst, const float* __restrict__ src, long n) {
  for (long i = (long)blockIdx.x * blockDim.x + threadIdx.x; i < n;
       i += (long)gridDim.x * blockDim.x)
    dst[i] = (_Float16)src[i];
}

// dst[n, 0:K] = a[n, :] ; dst[n, K:2K] = b[n, :]   (dst rows 2K wide)
__global__ void pack2_f16_kernel(_Float16* __restrict__ dst, const float* __restrict__ a,
                                 const float* __restrict__ b, long N, long K) {
  const long total = N * 2 * K;
  for (long i = (long)blockIdx.x * blockDim.x + threadIdx.x; i < total;
       i += (long)gridDim.x * blockDim.x) {
    const long row = i / (2 * K);
    const long j   = i - row * 2 * K;
    const float v  = (j < K) ? a[row * K + j] : b[row * K + (j - K)];
    dst[i] = (_Float16)v;
  }
}

__global__ void init_h_kernel(const float* __restrict__ h0,
                              float* __restrict__ hf0, float* __restrict__ hf1,
                              _Float16* __restrict__ h16_0, _Float16* __restrict__ h16_1) {
  const long n = (long)GB * GL * GH;
  for (long i = (long)blockIdx.x * blockDim.x + threadIdx.x; i < n;
       i += (long)gridDim.x * blockDim.x) {
    const int b = (int)(i / (GL * GH));
    const int l = (int)((i / GH) % GL);
    const int j = (int)(i % GH);
    const float v = h0[i];
    if (l == 0) { hf0[b * GH + j] = v; h16_0[b * GH + j] = (_Float16)v; }
    else        { hf1[b * GH + j] = v; h16_1[b * GH + j] = (_Float16)v; }
  }
}

__global__ void final_h_kernel(const float* __restrict__ hf0, const float* __restrict__ hf1,
                               float* __restrict__ out) {
  const long n = (long)GB * GL * GH;
  for (long i = (long)blockIdx.x * blockDim.x + threadIdx.x; i < n;
       i += (long)gridDim.x * blockDim.x) {
    const int b = (int)(i / (GL * GH));
    const int l = (int)((i / GH) % GL);
    const int j = (int)(i % GH);
    out[i] = (l == 0) ? hf0[b * GH + j] : hf1[b * GH + j];
  }
}

// ---------------------------------------------------------------------------
extern "C" void kernel_launch(void* const* d_in, const int* in_sizes, int n_in,
                              void* d_out, int out_size, void* d_ws, size_t ws_size,
                              hipStream_t stream) {
  (void)in_sizes; (void)n_in; (void)out_size; (void)ws_size;

  const float* x   = (const float*)d_in[0];
  const float* h0  = (const float*)d_in[1];
  const float* Wzx = (const float*)d_in[2];
  const float* bz  = (const float*)d_in[3];
  const float* Wzh = (const float*)d_in[4];
  const float* Wrx = (const float*)d_in[5];
  const float* br  = (const float*)d_in[6];
  const float* Wrh = (const float*)d_in[7];
  const float* Wgx = (const float*)d_in[8];
  const float* bg  = (const float*)d_in[9];
  const float* Wgh = (const float*)d_in[10];
  const float* Wo  = (const float*)d_in[11];
  const float* bo  = (const float*)d_in[12];

  // ---- carve workspace (all 256B aligned) ----
  size_t off = 0;
  auto carve = [&](size_t bytes) -> char* {
    char* p = (char*)d_ws + off;
    off += (bytes + 255) & ~(size_t)255;
    return p;
  };
  _Float16* x16    = (_Float16*)carve((size_t)GB * GS * GI * 2);          // 32 MB
  _Float16* Wzr0   = (_Float16*)carve((size_t)2 * GH * 2 * GH * 2);      //  8 MB
  _Float16* Wzr1   = (_Float16*)carve((size_t)2 * GH * 2 * GH * 2);      //  8 MB
  _Float16* Wg0    = (_Float16*)carve((size_t)GH * 2 * GH * 2);          //  4 MB
  _Float16* Wg1    = (_Float16*)carve((size_t)GH * 2 * GH * 2);          //  4 MB
  _Float16* Wo16   = (_Float16*)carve((size_t)GO * GH * 2);              //  2 MB
  _Float16* h16_0  = (_Float16*)carve((size_t)GB * GH * 2);
  _Float16* h16_1  = (_Float16*)carve((size_t)GB * GH * 2);
  _Float16* rh16   = (_Float16*)carve((size_t)GB * GH * 2);
  float*    hf0    = (float*)   carve((size_t)GB * GH * 4);
  float*    hf1    = (float*)   carve((size_t)GB * GH * 4);
  float*    Zbuf   = (float*)   carve((size_t)GB * GH * 4);

  // ---- one-time conversions / weight packing ----
  conv_f16_kernel<<<4096, 256, 0, stream>>>(x16, x, (long)GB * GS * GI);
  conv_f16_kernel<<<2048, 256, 0, stream>>>(Wo16, Wo, (long)GO * GH);
  for (int k = 0; k < GL; ++k) {
    _Float16* Wzr = (k == 0) ? Wzr0 : Wzr1;
    _Float16* Wg  = (k == 0) ? Wg0  : Wg1;
    const long wk = (long)k * GH * GH;   // I == H, so same offset for x/h weights
    pack2_f16_kernel<<<2048, 256, 0, stream>>>(Wzr,                       Wzx + wk, Wzh + wk, GH, GH);
    pack2_f16_kernel<<<2048, 256, 0, stream>>>(Wzr + (size_t)GH * 2 * GH, Wrx + wk, Wrh + wk, GH, GH);
    pack2_f16_kernel<<<2048, 256, 0, stream>>>(Wg,                        Wgx + wk, Wgh + wk, GH, GH);
  }
  init_h_kernel<<<256, 256, 0, stream>>>(h0, hf0, hf1, h16_0, h16_1);

  float* y = (float*)d_out;
  const long ldx = (long)GS * GI;   // batch-row stride inside x16[B,S,I]

  // ---- sequential scan over time ----
  for (int t = 0; t < GS; ++t) {
    const _Float16* x_t = x16 + (long)t * GI;
    // layer 0 (input = x[:, t, :])
    gru_zr_kernel<<<16, 128, 0, stream>>>(x_t, ldx, h16_0, Wzr0, bz, br, hf0, Zbuf, rh16);
    gru_g_kernel <<< 8, 128, 0, stream>>>(x_t, ldx, rh16, Wg0, bg, Zbuf, hf0, h16_0);
    // layer 1 (input = layer 0 new hidden)
    gru_zr_kernel<<<16, 128, 0, stream>>>(h16_0, (long)GH, h16_1, Wzr1,
                                          bz + GH, br + GH, hf1, Zbuf, rh16);
    gru_g_kernel <<< 8, 128, 0, stream>>>(h16_0, (long)GH, rh16, Wg1,
                                          bg + GH, Zbuf, hf1, h16_1);
    // output projection for this timestep
    gru_out_kernel<<< 8, 128, 0, stream>>>(h16_1, Wo16, bo, y, t);
  }

  // ---- final hidden state -> d_out tail ----
  final_h_kernel<<<256, 256, 0, stream>>>(hf0, hf1, y + (size_t)GB * GS * GO);
}
